// MoELoRA_83829171683650
// MI455X (gfx1250) — matrix operand order repaired and checked by reading, not compiled, not run on hardware
//
#include <hip/hip_runtime.h>
#include <hip/hip_bf16.h>

// ---------------- problem constants ----------------
#define TOKENS 8192
#define IN_F   4096
#define OUT_F  4096
#define RANK   16
#define N_EXP  8
#define KLORA  (N_EXP * RANK)   // 128
#define SCALING (1.0f / 16.0f)

// ---------------- GEMM tiling ----------------
#define BM 128
#define BN 128
#define BK 32
#define LDT  40    // padded LDS row stride in bf16 (16 dwords data + 4 dwords pad -> TDM pad_enable)
#define ABYTES (BM * LDT * 2)   // one LDS tile buffer in bytes

typedef __bf16 v16bf __attribute__((ext_vector_type(16)));
typedef __bf16 v8bf  __attribute__((ext_vector_type(8)));
typedef float  v8f   __attribute__((ext_vector_type(8)));
typedef unsigned int uint32x4 __attribute__((ext_vector_type(4)));
typedef int          int32x8  __attribute__((ext_vector_type(8)));
typedef int          int32x4  __attribute__((ext_vector_type(4)));

// ---- fp32 -> bf16 (round-to-nearest-even) ----
__device__ __forceinline__ unsigned short f2bf(float a) {
  unsigned u = __builtin_bit_cast(unsigned, a);
  return (unsigned short)((u + 0x7FFFu + ((u >> 16) & 1u)) >> 16);
}
__device__ __forceinline__ unsigned f2bf2(float a, float b) {
  unsigned ua = __builtin_bit_cast(unsigned, a);
  unsigned ub = __builtin_bit_cast(unsigned, b);
  ua = (ua + 0x7FFFu + ((ua >> 16) & 1u)) >> 16;
  ub = (ub + 0x7FFFu + ((ub >> 16) & 1u)) & 0xFFFF0000u;
  return (ua & 0xFFFFu) | ub;
}

// ---- TDM: 2-D tile (tile_rows x tile_k bf16) global -> LDS, with LDS padding ----
// D# per CDNA5 ISA ch8: group0 = {count/type/addr}, group1 = {sizes/strides/pad}
__device__ __forceinline__ void tdm_load_2d(unsigned lds_addr, const void* gptr,
                                            int tile_k, int tile_rows, int row_stride) {
  unsigned long long ga = (unsigned long long)(size_t)gptr;
  uint32x4 g0 = { 1u,                                   // count=1, is_restore=0, gather off
                  lds_addr,                             // lds_addr [63:32]
                  (unsigned)ga,                         // global_addr[31:0]
                  (unsigned)((ga >> 32) & 0x01FFFFFFu) | 0x80000000u }; // ga[56:32] | type=2
  const unsigned td0 = (unsigned)row_stride;            // tensor_dim0 (elements)
  const unsigned td1 = 0x00100000u;                     // tensor_dim1 (large, no OOB)
  int32x8 g1;
  g1[0] = (1 << 16)      // data_size = 2 bytes (bf16)
        | (1 << 20)      // pad_enable
        | (3 << 22)      // pad_interval: every 16 dwords (one 32-elem row)
        | (3 << 25);     // pad_amount: 4 dwords  -> LDS row stride = 40 bf16 = LDT
  g1[1] = (int)((td0 & 0xFFFFu) << 16);                               // tensor_dim0[15:0]
  g1[2] = (int)(((td0 >> 16) & 0xFFFFu) | ((td1 & 0xFFFFu) << 16));   // td0 hi | td1 lo
  g1[3] = (int)(((td1 >> 16) & 0xFFFFu) | ((unsigned)tile_k << 16));  // td1 hi | tile_dim0
  g1[4] = tile_rows;                                                  // tile_dim1 (tile_dim2=0)
  g1[5] = row_stride;                                                 // tensor_dim0_stride lo
  g1[6] = 0;
  g1[7] = 0;
  int32x4 z4 = { 0, 0, 0, 0 };
  int32x8 z8 = { 0, 0, 0, 0, 0, 0, 0, 0 };
  __builtin_amdgcn_tensor_load_to_lds(g0, g1, z4, z4, z8, 0);
}

// ---- fragment loads from LDS (ISA 7.12.2 layouts, two ds_load_b128 each) ----
__device__ __forceinline__ v16bf frag_A(const unsigned short* s, int row0, int lane) {
  int r = row0 + (lane & 15);
  int h = (lane >> 4) << 3;
  v8bf lo = *(const v8bf*)(s + r * LDT + h);
  v8bf hi = *(const v8bf*)(s + r * LDT + 16 + h);
  return __builtin_shufflevector(lo, hi, 0,1,2,3,4,5,6,7,8,9,10,11,12,13,14,15);
}
__device__ __forceinline__ v16bf frag_B(const unsigned short* s, int col0, int lane) {
  int n  = col0 + (lane & 15);
  int kb = (lane >> 4) << 3;
  v8bf lo = *(const v8bf*)(s + n * LDT + kb);
  v8bf hi = *(const v8bf*)(s + n * LDT + kb + 16);
  return __builtin_shufflevector(lo, hi, 0,1,2,3,4,5,6,7,8,9,10,11,12,13,14,15);
}

// ---- one K-step: 2x4 tiles of v_wmma_f32_16x16x32_bf16 per wave ----
__device__ __forceinline__ void compute_step(v8f acc[2][4], const unsigned short* sa,
                                             const unsigned short* sb, int wm, int wn, int lane) {
  v16bf a[2], b[4];
#pragma unroll
  for (int mi = 0; mi < 2; ++mi) a[mi] = frag_A(sa, wm * 32 + mi * 16, lane);
#pragma unroll
  for (int ni = 0; ni < 4; ++ni) b[ni] = frag_B(sb, wn * 64 + ni * 16, lane);
#pragma unroll
  for (int mi = 0; mi < 2; ++mi)
#pragma unroll
    for (int ni = 0; ni < 4; ++ni)
      acc[mi][ni] = __builtin_amdgcn_wmma_f32_16x16x32_bf16(
          false, a[mi], false, b[ni], (short)0, acc[mi][ni], false, false);
}

// =================== conversion kernels (fp32 -> bf16 workspace) ===================
__global__ __launch_bounds__(256) void conv_bf16(const float* __restrict__ src,
                                                 unsigned short* __restrict__ dst) {
  size_t i = (size_t)blockIdx.x * blockDim.x + threadIdx.x;   // pair index
  float2 f = *(const float2*)(src + 2 * i);
  ((unsigned*)dst)[i] = f2bf2(f.x, f.y);
}
// Vcat[n][e*16+r] = bf16(V[e][n][r])
__global__ __launch_bounds__(256) void conv_V(const float* __restrict__ V,
                                              unsigned short* __restrict__ vb) {
  int i = blockIdx.x * blockDim.x + threadIdx.x;              // pair index over [4096][64]
  int n = i >> 6, c = i & 63;
  int kg = 2 * c, e = kg >> 4, r = kg & 15;
  float2 f = *(const float2*)(V + (size_t)e * OUT_F * RANK + (size_t)n * RANK + r);
  ((unsigned*)vb)[i] = f2bf2(f.x, f.y);
}

// =================== K1: gating softmax ===================
__global__ __launch_bounds__(256) void moelora_gate(
    const float* __restrict__ x, const float* __restrict__ gW,
    const float* __restrict__ gb, float* __restrict__ gate) {
  const int lane = threadIdx.x & 31;
  const int wid  = threadIdx.x >> 5;
  const int b    = blockIdx.x * 8 + wid;
  const float* xr = x + (size_t)b * IN_F;

  float p[N_EXP];
#pragma unroll
  for (int e = 0; e < N_EXP; ++e) p[e] = 0.f;
  for (int k = lane; k < IN_F; k += 32) {
    float xv = xr[k];
#pragma unroll
    for (int e = 0; e < N_EXP; ++e) p[e] += xv * gW[e * IN_F + k];
  }
#pragma unroll
  for (int e = 0; e < N_EXP; ++e) {
#pragma unroll
    for (int off = 16; off > 0; off >>= 1) p[e] += __shfl_xor(p[e], off, 32);
    p[e] += gb[e];
  }
  float m = p[0];
#pragma unroll
  for (int e = 1; e < N_EXP; ++e) m = fmaxf(m, p[e]);
  float s = 0.f;
#pragma unroll
  for (int e = 0; e < N_EXP; ++e) { p[e] = __expf(p[e] - m); s += p[e]; }
  float inv = 1.f / s;
  float g = p[0];
#pragma unroll
  for (int e = 1; e < N_EXP; ++e) if (lane == e) g = p[e];
  if (lane < N_EXP) gate[(size_t)b * N_EXP + lane] = g * inv;
}

// =================== K2: g_h = bf16( gate*SCALING * (x @ U^T) ), N = 128 ===================
__global__ __launch_bounds__(256) void moelora_h(
    const unsigned short* __restrict__ xb, const unsigned short* __restrict__ ub,
    const float* __restrict__ gate, unsigned short* __restrict__ gh) {
  __shared__ __align__(16) unsigned short sA[2 * BM * LDT];
  __shared__ __align__(16) unsigned short sB[2 * BN * LDT];
  const int tid = threadIdx.x, lane = tid & 31, wid = tid >> 5;
  const int wm = wid >> 1, wn = wid & 1;
  const int M0 = blockIdx.y * BM;
  const unsigned ldsA = (unsigned)(size_t)(void*)sA;
  const unsigned ldsB = (unsigned)(size_t)(void*)sB;

  v8f acc[2][4];
#pragma unroll
  for (int mi = 0; mi < 2; ++mi)
#pragma unroll
    for (int ni = 0; ni < 4; ++ni) acc[mi][ni] = (v8f)0.f;

  if (wid == 0) tdm_load_2d(ldsA, xb + (size_t)M0 * IN_F, BK, BM, IN_F);
  if (wid == 1) tdm_load_2d(ldsB, ub, BK, BN, IN_F);
  if (wid < 2) __builtin_amdgcn_s_wait_tensorcnt(0);
  __syncthreads();

  const int NT = IN_F / BK;
#pragma unroll 1
  for (int kt = 0; kt < NT; ++kt) {
    const int cur = kt & 1, nxt = cur ^ 1;
    if (kt + 1 < NT) {
      if (wid == 0) tdm_load_2d(ldsA + nxt * ABYTES,
                                xb + (size_t)M0 * IN_F + (kt + 1) * BK, BK, BM, IN_F);
      if (wid == 1) tdm_load_2d(ldsB + nxt * ABYTES,
                                ub + (size_t)(kt + 1) * BK, BK, BN, IN_F);
    }
    compute_step(acc, sA + cur * (BM * LDT), sB + cur * (BN * LDT), wm, wn, lane);
    if (wid < 2) __builtin_amdgcn_s_wait_tensorcnt(0);
    __syncthreads();
  }

#pragma unroll
  for (int mi = 0; mi < 2; ++mi)
#pragma unroll
    for (int ni = 0; ni < 4; ++ni) {
      int col = wn * 64 + ni * 16 + (lane & 15);
      int rb  = M0 + wm * 32 + mi * 16 + ((lane >> 4) << 3);
      int e   = col >> 4;
#pragma unroll
      for (int v = 0; v < 8; ++v) {
        int row = rb + v;
        float s = gate[(size_t)row * N_EXP + e] * SCALING;
        gh[(size_t)row * KLORA + col] = f2bf(acc[mi][ni][v] * s);
      }
    }
}

// =================== K4: out = x@W^T + bias + g_h @ Vcat ===================
__global__ __launch_bounds__(256) void moelora_main(
    const unsigned short* __restrict__ xb, const unsigned short* __restrict__ wb,
    const float* __restrict__ bias, const unsigned short* __restrict__ vb,
    const unsigned short* __restrict__ gh, float* __restrict__ out) {
  __shared__ __align__(16) unsigned short sA[2 * BM * LDT];
  __shared__ __align__(16) unsigned short sB[2 * BN * LDT];
  const int tid = threadIdx.x, lane = tid & 31, wid = tid >> 5;
  const int wm = wid >> 1, wn = wid & 1;
  const int M0 = blockIdx.y * BM;
  const int N0 = blockIdx.x * BN;
  const unsigned ldsA = (unsigned)(size_t)(void*)sA;
  const unsigned ldsB = (unsigned)(size_t)(void*)sB;

  v8f acc[2][4];
#pragma unroll
  for (int mi = 0; mi < 2; ++mi)
#pragma unroll
    for (int ni = 0; ni < 4; ++ni) acc[mi][ni] = (v8f)0.f;

  // ---- LoRA accumulation: K = 128 over (g_h, Vcat), single-buffered ----
#pragma unroll 1
  for (int lt = 0; lt < KLORA / BK; ++lt) {
    if (wid == 0) tdm_load_2d(ldsA, gh + (size_t)M0 * KLORA + lt * BK, BK, BM, KLORA);
    if (wid == 1) tdm_load_2d(ldsB, vb + (size_t)N0 * KLORA + lt * BK, BK, BN, KLORA);
    if (wid < 2) __builtin_amdgcn_s_wait_tensorcnt(0);
    __syncthreads();
    compute_step(acc, sA, sB, wm, wn, lane);
    __syncthreads();
  }

  // ---- main GEMM: K = 4096 over (x_bf16, W_bf16), TDM double-buffered ----
  if (wid == 0) tdm_load_2d(ldsA, xb + (size_t)M0 * IN_F, BK, BM, IN_F);
  if (wid == 1) tdm_load_2d(ldsB, wb + (size_t)N0 * IN_F, BK, BN, IN_F);
  if (wid < 2) __builtin_amdgcn_s_wait_tensorcnt(0);
  __syncthreads();

  const int NT = IN_F / BK;
#pragma unroll 1
  for (int kt = 0; kt < NT; ++kt) {
    const int cur = kt & 1, nxt = cur ^ 1;
    if (kt + 1 < NT) {
      if (wid == 0) tdm_load_2d(ldsA + nxt * ABYTES,
                                xb + (size_t)M0 * IN_F + (kt + 1) * BK, BK, BM, IN_F);
      if (wid == 1) tdm_load_2d(ldsB + nxt * ABYTES,
                                wb + (size_t)N0 * IN_F + (kt + 1) * BK, BK, BN, IN_F);
    }
    compute_step(acc, sA + cur * (BM * LDT), sB + cur * (BN * LDT), wm, wn, lane);
    if (wid < 2) __builtin_amdgcn_s_wait_tensorcnt(0);
    __syncthreads();
  }

  // ---- epilogue: + bias, store ----
#pragma unroll
  for (int mi = 0; mi < 2; ++mi)
#pragma unroll
    for (int ni = 0; ni < 4; ++ni) {
      int col = N0 + wn * 64 + ni * 16 + (lane & 15);
      int rb  = M0 + wm * 32 + mi * 16 + ((lane >> 4) << 3);
      float bc = bias[col];
#pragma unroll
      for (int v = 0; v < 8; ++v)
        out[(size_t)(rb + v) * OUT_F + col] = acc[mi][ni][v] + bc;
    }
}

extern "C" void kernel_launch(void* const* d_in, const int* in_sizes, int n_in,
                              void* d_out, int out_size, void* d_ws, size_t ws_size,
                              hipStream_t stream) {
  const float* x    = (const float*)d_in[0];
  const float* w    = (const float*)d_in[1];
  const float* bias = (const float*)d_in[2];
  const float* gW   = (const float*)d_in[3];
  const float* gb   = (const float*)d_in[4];
  const float* U    = (const float*)d_in[5];
  const float* V    = (const float*)d_in[6];
  float* out = (float*)d_out;

  // ---- workspace layout (all 256B-aligned) ----
  char* ws = (char*)d_ws;
  float*          gate = (float*)ws;                       ws += (size_t)TOKENS * N_EXP * 4;   // 256 KB
  unsigned short* gh   = (unsigned short*)ws;              ws += (size_t)TOKENS * KLORA * 2;   // 2 MB
  unsigned short* xb   = (unsigned short*)ws;              ws += (size_t)TOKENS * IN_F * 2;    // 64 MB
  unsigned short* wb   = (unsigned short*)ws;              ws += (size_t)OUT_F * IN_F * 2;     // 32 MB
  unsigned short* ub   = (unsigned short*)ws;              ws += (size_t)KLORA * IN_F * 2;     // 1 MB
  unsigned short* vb   = (unsigned short*)ws;                                                  // 1 MB

  conv_bf16<<<(TOKENS * IN_F / 2) / 256, 256, 0, stream>>>(x, xb);
  conv_bf16<<<(OUT_F * IN_F / 2) / 256, 256, 0, stream>>>(w, wb);
  conv_bf16<<<(KLORA * IN_F / 2) / 256, 256, 0, stream>>>(U, ub);
  conv_V  <<<(OUT_F * KLORA / 2) / 256, 256, 0, stream>>>(V, vb);

  moelora_gate<<<TOKENS / 8, 256, 0, stream>>>(x, gW, gb, gate);
  moelora_h<<<dim3(1, TOKENS / BM), 256, 0, stream>>>(xb, ub, gate, gh);
  moelora_main<<<dim3(OUT_F / BN, TOKENS / BM), 256, 0, stream>>>(xb, wb, bias, vb, gh, out);
}
